// TreeLRU_87582973100343
// MI455X (gfx1250) — matrix operand order repaired and checked by reading, not compiled, not run on hardware
//
#include <hip/hip_runtime.h>
#include <hip/hip_bf16.h>

typedef _Float16 half4  __attribute__((ext_vector_type(4)));
typedef _Float16 half8  __attribute__((ext_vector_type(8)));
typedef _Float16 half16 __attribute__((ext_vector_type(16)));
typedef float    floatx8 __attribute__((ext_vector_type(8)));
typedef int      i32x4  __attribute__((ext_vector_type(4)));

#define AS1 __attribute__((address_space(1)))
#define AS3 __attribute__((address_space(3)))

#define IN_F     512
#define OUT_F    512
#define STATE_F  256
#define BATCH    16
#define N_NODES  4095
#define RTOT     (BATCH * N_NODES)   // 65520
#define KDIM     512
#define CDIM     512

// ---- gfx1250 async global->LDS path (probe via __has_builtin, sync fallback) ----
#if defined(__has_builtin)
# if __has_builtin(__builtin_amdgcn_global_load_async_to_lds_b128)
#  define USE_ASYNC_LDS 1
# endif
#endif
#ifndef USE_ASYNC_LDS
# define USE_ASYNC_LDS 0
#endif

#if USE_ASYNC_LDS
__device__ __forceinline__ void async_copy16(const _Float16* src, _Float16* dst) {
  __builtin_amdgcn_global_load_async_to_lds_b128(
      (AS1 i32x4*)(AS1 void*)(void*)const_cast<_Float16*>(src),
      (AS3 i32x4*)(AS3 void*)(void*)dst,
      0, 0);
}
#endif

__device__ __forceinline__ void wait_async_lds() {
#if USE_ASYNC_LDS
# if __has_builtin(__builtin_amdgcn_s_wait_asynccnt)
  __builtin_amdgcn_s_wait_asynccnt(0);
# else
  asm volatile("s_wait_asynccnt 0x0" ::: "memory");
# endif
#endif
}

// ---------------- prep: LRU constants ----------------
__global__ void prep_consts(const float* __restrict__ nu_log,
                            const float* __restrict__ theta_log,
                            const float* __restrict__ gamma_log,
                            float* __restrict__ lr, float* __restrict__ li,
                            float* __restrict__ gm) {
  int s = threadIdx.x;
  if (s < STATE_F) {
    float lam = expf(-expf(nu_log[s]));
    float th  = expf(theta_log[s]);
    lr[s] = lam * cosf(th);
    li[s] = lam * sinf(th);
    gm[s] = expf(gamma_log[s]);
  }
}

// ---------------- prep: Mcat = [B_re@W_in ; B_im@W_in] in f16, bias_in = B@b_in ----------------
__global__ void prep_m(const float* __restrict__ W_in,
                       const float* __restrict__ B_re,
                       const float* __restrict__ B_im,
                       const float* __restrict__ b_in,
                       _Float16* __restrict__ Mcat,
                       float* __restrict__ bias_in) {
  int linear = blockIdx.x * blockDim.x + threadIdx.x;   // 512*512 threads
  int k = linear & (KDIM - 1);
  int c = linear >> 9;
  const float* Brow = (c < STATE_F) ? (B_re + (size_t)c * IN_F)
                                    : (B_im + (size_t)(c - STATE_F) * IN_F);
  float acc = 0.f;
  for (int j = 0; j < IN_F; ++j)
    acc = fmaf(Brow[j], W_in[(size_t)j * IN_F + k], acc);
  Mcat[(size_t)c * KDIM + k] = (_Float16)acc;
  if (k == 0) {
    float bacc = 0.f;
    for (int j = 0; j < IN_F; ++j) bacc = fmaf(Brow[j], b_in[j], bacc);
    bias_in[c] = bacc;
  }
}

// ---------------- prep: Ccat = [C_re | -C_im] in f16, plus zero bias ----------------
__global__ void prep_c(const float* __restrict__ C_re,
                       const float* __restrict__ C_im,
                       _Float16* __restrict__ Ccat,
                       float* __restrict__ bias_zero) {
  int linear = blockIdx.x * blockDim.x + threadIdx.x;   // 512*512 threads
  int s = linear & (CDIM - 1);
  int o = linear >> 9;
  float v = (s < STATE_F) ? C_re[(size_t)o * STATE_F + s]
                          : -C_im[(size_t)o * STATE_F + (s - STATE_F)];
  Ccat[(size_t)o * CDIM + s] = (_Float16)v;
  if (linear < CDIM) bias_zero[linear] = 0.f;
}

// ---------------- main GEMM: Y[R,512] = f32(X[R,512]) * f16 Wt[512,512]^T + bias ----------------
// Block tile: 64 rows x 512 cols (A tile read exactly once), K-blocked at 64.
// 8 waves = 4 row-strips x 2 col-strips; each wave: 16 rows x 256 cols = 16 acc tiles.
#define BROW 64
#define KB   64
#define LDA  72   // halves per A row in LDS (144B: 16B aligned, bank-spread)
#define LDB  72   // halves per W row in LDS
#define SMEM_A_HALVES (BROW * LDA)     // 4608
#define SMEM_B_HALVES (CDIM * LDB)     // 36864
#define SMEM_BYTES    ((SMEM_A_HALVES + SMEM_B_HALVES) * 2)   // 82944

__global__ __launch_bounds__(256) void gemm_xw_f16(
    const float* __restrict__ X,        // [R, KDIM] f32
    const _Float16* __restrict__ Wt,    // [CDIM, KDIM] f16 row-major
    const float* __restrict__ bias,     // [CDIM]
    float* __restrict__ Y,              // [R, CDIM] f32
    int R) {
  extern __shared__ _Float16 smem[];
  _Float16* Alds = smem;
  _Float16* Blds = smem + SMEM_A_HALVES;

  const int tid  = threadIdx.x;
  const int lane = tid & 31;
  const int wave = tid >> 5;            // 0..7
  const int rowStrip = wave & 3;        // 0..3  -> rows [rowStrip*16, +16)
  const int colStrip = wave >> 2;       // 0..1  -> cols [colStrip*256, +256)
  const int l16  = lane & 15;
  const int kgrp = lane >> 4;           // 0/1 lane half-group
  const int rowBase = blockIdx.x * BROW;

  floatx8 acc[16];
#pragma unroll
  for (int t = 0; t < 16; ++t) {
    floatx8 z = {0.f, 0.f, 0.f, 0.f, 0.f, 0.f, 0.f, 0.f};
    acc[t] = z;
  }

  for (int kb = 0; kb < KDIM; kb += KB) {
    // ---- stage A tile: 64 rows x 64 k, f32 global -> f16 LDS ----
    // phase 1: issue all global loads (4 in flight per thread), phase 2: convert+store
    {
      const int kq = tid & 15;          // float4 chunk (4 k)
      const int r0 = tid >> 4;          // 0..15
      float4 tmp[4];
#pragma unroll
      for (int p = 0; p < 4; ++p) {
        const int rg = rowBase + r0 + p * 16;
        float4 v = make_float4(0.f, 0.f, 0.f, 0.f);
        if (rg < R)
          v = *reinterpret_cast<const float4*>(X + (size_t)rg * KDIM + kb + kq * 4);
        tmp[p] = v;
      }
#pragma unroll
      for (int p = 0; p < 4; ++p) {
        const int r = r0 + p * 16;
        half4 hv = {(_Float16)tmp[p].x, (_Float16)tmp[p].y,
                    (_Float16)tmp[p].z, (_Float16)tmp[p].w};
        *reinterpret_cast<half4*>(&Alds[r * LDA + kq * 4]) = hv;
      }
    }
    // ---- stage W tile: 512 cols x 64 k, f16 global -> f16 LDS ----
    {
      const int kc = tid & 7;           // 8-half (16B) chunk
      const int c0 = tid >> 3;          // 0..31
#pragma unroll
      for (int cc = 0; cc < CDIM; cc += 32) {
        const int c = c0 + cc;
        const _Float16* src = Wt + (size_t)c * KDIM + kb + kc * 8;
        _Float16* dst = &Blds[c * LDB + kc * 8];
#if USE_ASYNC_LDS
        async_copy16(src, dst);
#else
        *reinterpret_cast<half8*>(dst) = *reinterpret_cast<const half8*>(src);
#endif
      }
    }
    wait_async_lds();
    __syncthreads();

#pragma unroll
    for (int kk = 0; kk < KB; kk += 32) {
      // A fragment (16x32 f16): lane holds row lane&15; K split per ISA layout
      const int arow = rowStrip * 16 + l16;
      const half8 alo = *reinterpret_cast<const half8*>(&Alds[arow * LDA + kk + kgrp * 8]);
      const half8 ahi = *reinterpret_cast<const half8*>(&Alds[arow * LDA + kk + 16 + kgrp * 8]);
      const half16 a = __builtin_shufflevector(alo, ahi,
          0, 1, 2, 3, 4, 5, 6, 7, 8, 9, 10, 11, 12, 13, 14, 15);

      // B fragment loader (32x16 f16): lane holds col lane&15; K split per ISA layout (same as A)
      auto load_b = [&](int ct) -> half16 {
        const int c = colStrip * 256 + ct * 16 + l16;
        const half8 blo = *reinterpret_cast<const half8*>(&Blds[c * LDB + kk + kgrp * 8]);
        const half8 bhi = *reinterpret_cast<const half8*>(&Blds[c * LDB + kk + 16 + kgrp * 8]);
        return __builtin_shufflevector(blo, bhi,
            0, 1, 2, 3, 4, 5, 6, 7, 8, 9, 10, 11, 12, 13, 14, 15);
      };

      // 2-deep software pipeline: B load for ct+1 issues before WMMA for ct
      half16 bcur = load_b(0);
#pragma unroll
      for (int ct = 0; ct < 16; ++ct) {
        half16 bnext = bcur;
        if (ct < 15) bnext = load_b(ct + 1);
        acc[ct] = __builtin_amdgcn_wmma_f32_16x16x32_f16(
            false, a, false, bcur, (short)0, acc[ct], false, false);
        bcur = bnext;
      }
    }
    __syncthreads();
  }

  // ---- store: D layout -> VGPR i holds row (i + kgrp*8), col = lane&15 ----
#pragma unroll
  for (int ct = 0; ct < 16; ++ct) {
    const int colg = colStrip * 256 + ct * 16 + l16;
    const float bv = bias[colg];
#pragma unroll
    for (int i = 0; i < 8; ++i) {
      const int rg = rowBase + rowStrip * 16 + kgrp * 8 + i;
      if (rg < R) Y[(size_t)rg * CDIM + colg] = acc[ct][i] + bv;
    }
  }
}

// ---------------- tree recurrence: one level per launch ----------------
__global__ void tree_level(const float* __restrict__ inb,   // [R,512] (re|im)
                           float* __restrict__ h,           // [R,512] (re|im)
                           const float* __restrict__ lr,
                           const float* __restrict__ li,
                           const float* __restrict__ gm,
                           int levelBase, int nLevel, int hasChildren) {
  const int gtid = blockIdx.x * blockDim.x + threadIdx.x;
  const int s = gtid & (STATE_F - 1);
  const int t = gtid >> 8;
  if (t >= nLevel * BATCH) return;
  const int l    = t % nLevel;
  const int bidx = t / nLevel;
  const int node = levelBase + l;
  const size_t r = (size_t)bidx * N_NODES + node;

  const float inr = inb[r * 512 + s];
  const float ini = inb[r * 512 + 256 + s];
  float cr = 0.f, ci = 0.f;
  if (hasChildren) {
    const size_t rl = (size_t)bidx * N_NODES + (2 * node + 1);
    const size_t rr = rl + 1;
    cr = h[rl * 512 + s]       + h[rr * 512 + s];
    ci = h[rl * 512 + 256 + s] + h[rr * 512 + 256 + s];
  }
  const float Lr = lr[s], Li = li[s], g = gm[s];
  h[r * 512 + s]       = Lr * cr - Li * ci + g * inr;
  h[r * 512 + 256 + s] = Lr * ci + Li * cr + g * ini;
}

// ---------------- host orchestration ----------------
extern "C" void kernel_launch(void* const* d_in, const int* in_sizes, int n_in,
                              void* d_out, int out_size, void* d_ws, size_t ws_size,
                              hipStream_t stream) {
  const float* x      = (const float*)d_in[0];
  // d_in[1] = idx_batch (unused by the reference)
  const float* W_in   = (const float*)d_in[2];
  const float* b_in   = (const float*)d_in[3];
  const float* nu_log = (const float*)d_in[4];
  const float* th_log = (const float*)d_in[5];
  const float* gm_log = (const float*)d_in[6];
  const float* B_re   = (const float*)d_in[7];
  const float* B_im   = (const float*)d_in[8];
  const float* C_re   = (const float*)d_in[9];
  const float* C_im   = (const float*)d_in[10];
  float* out = (float*)d_out;

  // workspace carve-out (256B aligned)
  char* ws = (char*)d_ws;
  size_t off = 0;
  auto carve = [&](size_t bytes) -> void* {
    void* p = ws + off;
    off = (off + bytes + 255) & ~(size_t)255;
    return p;
  };
  float*     in_buf  = (float*)    carve((size_t)RTOT * 512 * sizeof(float));
  float*     h_buf   = (float*)    carve((size_t)RTOT * 512 * sizeof(float));
  _Float16*  Mcat    = (_Float16*) carve((size_t)CDIM * KDIM * sizeof(_Float16));
  _Float16*  Ccat    = (_Float16*) carve((size_t)CDIM * KDIM * sizeof(_Float16));
  float*     bias_in = (float*)    carve(CDIM * sizeof(float));
  float*     bias0   = (float*)    carve(CDIM * sizeof(float));
  float*     lr      = (float*)    carve(STATE_F * sizeof(float));
  float*     li      = (float*)    carve(STATE_F * sizeof(float));
  float*     gm      = (float*)    carve(STATE_F * sizeof(float));

  // 1. constants + folded weight matrices
  prep_consts<<<1, 256, 0, stream>>>(nu_log, th_log, gm_log, lr, li, gm);
  prep_m<<<(CDIM * KDIM) / 256, 256, 0, stream>>>(W_in, B_re, B_im, b_in, Mcat, bias_in);
  prep_c<<<(CDIM * KDIM) / 256, 256, 0, stream>>>(C_re, C_im, Ccat, bias0);

  // 2. input projection: in_buf = X @ Mcat^T + bias_in   (re | im concatenated)
  dim3 ggrid((RTOT + BROW - 1) / BROW);
  gemm_xw_f16<<<ggrid, 256, SMEM_BYTES, stream>>>(x, Mcat, bias_in, in_buf, RTOT);

  // 3. tree recurrence, bottom-up, one launch per level
  for (int d = 11; d >= 0; --d) {
    const int nLevel = 1 << d;
    const int levelBase = nLevel - 1;
    tree_level<<<BATCH * nLevel, 256, 0, stream>>>(
        in_buf, h_buf, lr, li, gm, levelBase, nLevel, (d < 11) ? 1 : 0);
  }

  // 4. output projection: out = hcat @ Ccat^T
  gemm_xw_f16<<<ggrid, 256, SMEM_BYTES, stream>>>(h_buf, Ccat, bias0, out, RTOT);
}
